// GraphRNNBilinear_32220844654752
// MI455X (gfx1250) — compile-verified, matrix operand order checked by reading, not run on hardware
//
#include <hip/hip_runtime.h>
#include <hip/hip_bf16.h>

typedef __bf16 bf16_t;
typedef __attribute__((ext_vector_type(16))) __bf16 v16bf;
typedef __attribute__((ext_vector_type(8)))  __bf16 v8bf;
typedef __attribute__((ext_vector_type(8)))  float  v8f;

#define B_   64
#define T_   511
#define S_   512
#define DE_  400
#define D_   500
#define DP_  512
#define HG_  384
#define HE_  64
#define DT_  100
#define DTP_ 112
#define BS_  (B_ * S_)
#define KTH_ (HG_ / 32)   // 12
#define NTG_ (HG_ / 16)   // 24

// ---------------- device helpers ----------------

__device__ __forceinline__ float sigf(float x) { return 1.0f / (1.0f + __expf(-x)); }
__device__ __forceinline__ float eluf(float x) { return x > 0.0f ? x : (__expf(x) - 1.0f); }

// A fragment (16x32 bf16) from row-major bf16. base points at (row0, col 0).
// ISA layout: lanes 0-15 row=lane, K halves {k0..k0+7, k0+16..k0+23};
// lanes 16-31 same rows, K offset +8.
__device__ __forceinline__ v16bf load_a_frag(const bf16_t* base, int row_stride, int k0, int lane) {
  const bf16_t* p = base + (size_t)(lane & 15) * row_stride + k0 + ((lane >> 4) << 3);
  v8bf lo = *(const v8bf*)(p);
  v8bf hi = *(const v8bf*)(p + 16);
  v16bf r;
#pragma unroll
  for (int h = 0; h < 8; ++h) { r[h] = lo[h]; r[8 + h] = hi[h]; }
  return r;
}

// B fragment from pre-swizzled weights: each (ntile,ktile) tile stores 32 lanes x 16 halves contiguous.
__device__ __forceinline__ v16bf load_w_frag(const bf16_t* wsw, int tile_idx, int lane) {
  const bf16_t* p = wsw + (size_t)tile_idx * 512 + (size_t)lane * 16;
  v8bf lo = *(const v8bf*)(p);
  v8bf hi = *(const v8bf*)(p + 8);
  v16bf r;
#pragma unroll
  for (int h = 0; h < 8; ++h) { r[h] = lo[h]; r[8 + h] = hi[h]; }
  return r;
}

__device__ __forceinline__ v8f wmma_bf16(v16bf a, v16bf b, v8f c) {
  return __builtin_amdgcn_wmma_f32_16x16x32_bf16(false, a, false, b, (short)0, c, false, false);
}

// ---------------- prep kernels ----------------

__global__ void build_x_kernel(const float* __restrict__ inp, const float* __restrict__ tag,
                               const float* __restrict__ sent, bf16_t* __restrict__ xbf) {
  size_t idx = (size_t)blockIdx.x * blockDim.x + threadIdx.x;
  size_t total = (size_t)B_ * S_ * DP_;
  if (idx >= total) return;
  int k = (int)(idx % DP_);
  size_t bs = idx / DP_;
  int s = (int)(bs % S_);
  int b = (int)(bs / S_);
  float v = 0.0f;
  if (k < D_) {
    if (s == 0) v = sent[k];
    else if (k < DE_) v = inp[((size_t)b * T_ + (s - 1)) * DE_ + k];
    else v = tag[((size_t)b * T_ + (s - 1)) * DT_ + (k - DE_)];
  }
  xbf[idx] = (bf16_t)v;
}

// Swizzle W [N,K] fp32 row-major into B-fragment tiles of W^T (KxN), bf16, zero padded.
__global__ void swizzle_w_kernel(const float* __restrict__ W, bf16_t* __restrict__ out,
                                 int N, int Npad, int K, int KT) {
  size_t idx = (size_t)blockIdx.x * blockDim.x + threadIdx.x;
  size_t total = (size_t)(Npad / 16) * KT * 512;
  if (idx >= total) return;
  int h = (int)(idx & 15);
  int lane = (int)((idx >> 4) & 31);
  size_t tile = idx >> 9;
  int kt = (int)(tile % KT);
  int nt = (int)(tile / KT);
  int k = kt * 32 + (h & 7) + ((h >> 3) << 4) + ((lane >> 4) << 3);
  int n = nt * 16 + (lane & 15);
  float v = (k < K && n < N) ? W[(size_t)n * K + k] : 0.0f;
  out[idx] = (bf16_t)v;
}

// ---------------- graph LSTM step: fused GEMM + cell ----------------
// grid = (24, 2), block 32. Wave owns 32 batch rows x 16 hidden cols x 4 gates
// (8 accumulator tiles). Per K-step: 6 loads hoisted into one clause, then
// 8 back-to-back WMMAs. KTA is compile-time so the scheduler can overlap.
template <int KTA>
__global__ void graph_step_kernel(const bf16_t* __restrict__ Abase, int a_stride,
                                  const bf16_t* __restrict__ Wih, const bf16_t* __restrict__ Whh,
                                  const float* __restrict__ bih, const float* __restrict__ bhh,
                                  const bf16_t* __restrict__ hprev, bf16_t* __restrict__ hnext,
                                  float* __restrict__ cbuf,
                                  bf16_t* __restrict__ outH, int out_stride,
                                  const int* __restrict__ maskp, int t) {
  const int lane = threadIdx.x;
  const int nt = blockIdx.x;        // hidden tile 0..23
  const int mb = blockIdx.y;        // batch half 0..1 (32 rows)

  v8f acc[2][4];
#pragma unroll
  for (int m = 0; m < 2; ++m)
#pragma unroll
    for (int g = 0; g < 4; ++g) acc[m][g] = (v8f){};

  const bf16_t* A0 = Abase + (size_t)(mb * 32) * a_stride;
  const bf16_t* A1 = A0 + (size_t)16 * a_stride;
  const bf16_t* H0r = hprev + (size_t)(mb * 32) * HG_;
  const bf16_t* H1r = H0r + (size_t)16 * HG_;

  // phase 1: input GEMM
  for (int kt = 0; kt < KTA; ++kt) {
    v16bf a0 = load_a_frag(A0, a_stride, kt * 32, lane);
    v16bf a1 = load_a_frag(A1, a_stride, kt * 32, lane);
    v16bf b0 = load_w_frag(Wih, (nt + 0 * NTG_) * KTA + kt, lane);
    v16bf b1 = load_w_frag(Wih, (nt + 1 * NTG_) * KTA + kt, lane);
    v16bf b2 = load_w_frag(Wih, (nt + 2 * NTG_) * KTA + kt, lane);
    v16bf b3 = load_w_frag(Wih, (nt + 3 * NTG_) * KTA + kt, lane);
    acc[0][0] = wmma_bf16(a0, b0, acc[0][0]);
    acc[1][0] = wmma_bf16(a1, b0, acc[1][0]);
    acc[0][1] = wmma_bf16(a0, b1, acc[0][1]);
    acc[1][1] = wmma_bf16(a1, b1, acc[1][1]);
    acc[0][2] = wmma_bf16(a0, b2, acc[0][2]);
    acc[1][2] = wmma_bf16(a1, b2, acc[1][2]);
    acc[0][3] = wmma_bf16(a0, b3, acc[0][3]);
    acc[1][3] = wmma_bf16(a1, b3, acc[1][3]);
  }
  // phase 2: recurrent GEMM
  for (int kt = 0; kt < KTH_; ++kt) {
    v16bf a0 = load_a_frag(H0r, HG_, kt * 32, lane);
    v16bf a1 = load_a_frag(H1r, HG_, kt * 32, lane);
    v16bf b0 = load_w_frag(Whh, (nt + 0 * NTG_) * KTH_ + kt, lane);
    v16bf b1 = load_w_frag(Whh, (nt + 1 * NTG_) * KTH_ + kt, lane);
    v16bf b2 = load_w_frag(Whh, (nt + 2 * NTG_) * KTH_ + kt, lane);
    v16bf b3 = load_w_frag(Whh, (nt + 3 * NTG_) * KTH_ + kt, lane);
    acc[0][0] = wmma_bf16(a0, b0, acc[0][0]);
    acc[1][0] = wmma_bf16(a1, b0, acc[1][0]);
    acc[0][1] = wmma_bf16(a0, b1, acc[0][1]);
    acc[1][1] = wmma_bf16(a1, b1, acc[1][1]);
    acc[0][2] = wmma_bf16(a0, b2, acc[0][2]);
    acc[1][2] = wmma_bf16(a1, b2, acc[1][2]);
    acc[0][3] = wmma_bf16(a0, b3, acc[0][3]);
    acc[1][3] = wmma_bf16(a1, b3, acc[1][3]);
  }

  const int col = nt * 16 + (lane & 15);
  const float b0 = bih[col] + bhh[col];
  const float b1 = bih[HG_ + col] + bhh[HG_ + col];
  const float b2 = bih[2 * HG_ + col] + bhh[2 * HG_ + col];
  const float b3 = bih[3 * HG_ + col] + bhh[3 * HG_ + col];

#pragma unroll
  for (int m = 0; m < 2; ++m) {
#pragma unroll
    for (int i = 0; i < 8; ++i) {
      const int row = mb * 32 + m * 16 + i + ((lane >> 4) << 3);
      const float gi = sigf(acc[m][0][i] + b0);
      const float gf = sigf(acc[m][1][i] + b1);
      const float gg = tanhf(acc[m][2][i] + b2);
      const float go = sigf(acc[m][3][i] + b3);
      const size_t ci = (size_t)row * HG_ + col;
      const float c = gf * cbuf[ci] + gi * gg;
      cbuf[ci] = c;
      const float h = go * tanhf(c);
      hnext[ci] = (bf16_t)h;
      float mkv = 1.0f;
      if (maskp) mkv = (t == 0) ? 1.0f : (float)maskp[row * T_ + (t - 1)];
      outH[(size_t)row * out_stride + col] = (bf16_t)(h * mkv);
    }
  }
}

// ---------------- g2e projection: init edge LSTM h-state, zero c ----------------
// grid = (4, BS/32), block 32. 2 M-tiles per wave, hoisted loads.
__global__ void proj_kernel(const bf16_t* __restrict__ G, const bf16_t* __restrict__ Wsw,
                            const float* __restrict__ bias,
                            bf16_t* __restrict__ h0, bf16_t* __restrict__ h1,
                            float* __restrict__ c0, float* __restrict__ c1) {
  const int lane = threadIdx.x;
  const int nt = blockIdx.x;
  const int mb = blockIdx.y;
  v8f acc0 = {}, acc1 = {};
  const bf16_t* A0 = G + (size_t)mb * 32 * HG_;
  const bf16_t* A1 = A0 + (size_t)16 * HG_;
#pragma unroll
  for (int kt = 0; kt < KTH_; ++kt) {
    v16bf a0 = load_a_frag(A0, HG_, kt * 32, lane);
    v16bf a1 = load_a_frag(A1, HG_, kt * 32, lane);
    v16bf b = load_w_frag(Wsw, nt * KTH_ + kt, lane);
    acc0 = wmma_bf16(a0, b, acc0);
    acc1 = wmma_bf16(a1, b, acc1);
  }
  const int col = nt * 16 + (lane & 15);
  const float bv = bias[col];
#pragma unroll
  for (int i = 0; i < 8; ++i) {
    const size_t r0 = (size_t)mb * 32 + i + ((lane >> 4) << 3);
    const size_t r1 = r0 + 16;
    const float v0 = acc0[i] + bv, v1 = acc1[i] + bv;
    const size_t i0 = r0 * HE_ + col, i1 = r1 * HE_ + col;
    h0[i0] = (bf16_t)v0; h1[i0] = (bf16_t)v0; c0[i0] = 0.0f; c1[i0] = 0.0f;
    h0[i1] = (bf16_t)v1; h1[i1] = (bf16_t)v1; c0[i1] = 0.0f; c1[i1] = 0.0f;
  }
}

// ---------------- tag heads: elu(G @ W^T + b), N=100 (padded 112) ----------------
// grid = (7, BS/32), block 32.
__global__ void tag_kernel(const bf16_t* __restrict__ G, const bf16_t* __restrict__ Wsw,
                           const float* __restrict__ bias, float* __restrict__ out) {
  const int lane = threadIdx.x;
  const int nt = blockIdx.x;
  const int mb = blockIdx.y;
  v8f acc0 = {}, acc1 = {};
  const bf16_t* A0 = G + (size_t)mb * 32 * HG_;
  const bf16_t* A1 = A0 + (size_t)16 * HG_;
#pragma unroll
  for (int kt = 0; kt < KTH_; ++kt) {
    v16bf a0 = load_a_frag(A0, HG_, kt * 32, lane);
    v16bf a1 = load_a_frag(A1, HG_, kt * 32, lane);
    v16bf b = load_w_frag(Wsw, nt * KTH_ + kt, lane);
    acc0 = wmma_bf16(a0, b, acc0);
    acc1 = wmma_bf16(a1, b, acc1);
  }
  const int col = nt * 16 + (lane & 15);
  if (col < DT_) {
    const float bv = bias[col];
#pragma unroll
    for (int i = 0; i < 8; ++i) {
      const size_t r0 = (size_t)mb * 32 + i + ((lane >> 4) << 3);
      out[r0 * DT_ + col] = eluf(acc0[i] + bv);
      out[(r0 + 16) * DT_ + col] = eluf(acc1[i] + bv);
    }
  }
}

// ---------------- edge LSTM layer 0 step ----------------
// grid = (4, BS/32), block 32. Scalar band input folded elementwise via Wih0 [256].
__global__ void edge0_step_kernel(const bf16_t* __restrict__ hprev, bf16_t* __restrict__ hnext,
                                  float* __restrict__ cbuf,
                                  const float* __restrict__ Wih0,
                                  const float* __restrict__ bih, const float* __restrict__ bhh,
                                  const bf16_t* __restrict__ WhhSw,
                                  const int* __restrict__ heads, const float* __restrict__ bos,
                                  int s) {
  const int lane = threadIdx.x;
  const int nt = blockIdx.x;    // 0..3
  const int mb = blockIdx.y;    // 0..1023
  v8f acc[2][4];
#pragma unroll
  for (int m = 0; m < 2; ++m)
#pragma unroll
    for (int g = 0; g < 4; ++g) acc[m][g] = (v8f){};
  const bf16_t* H0r = hprev + (size_t)mb * 32 * HE_;
  const bf16_t* H1r = H0r + (size_t)16 * HE_;
#pragma unroll
  for (int kt = 0; kt < 2; ++kt) {
    v16bf a0 = load_a_frag(H0r, HE_, kt * 32, lane);
    v16bf a1 = load_a_frag(H1r, HE_, kt * 32, lane);
    v16bf b0 = load_w_frag(WhhSw, (nt + 0) * 2 + kt, lane);
    v16bf b1 = load_w_frag(WhhSw, (nt + 4) * 2 + kt, lane);
    v16bf b2 = load_w_frag(WhhSw, (nt + 8) * 2 + kt, lane);
    v16bf b3 = load_w_frag(WhhSw, (nt + 12) * 2 + kt, lane);
    acc[0][0] = wmma_bf16(a0, b0, acc[0][0]);
    acc[1][0] = wmma_bf16(a1, b0, acc[1][0]);
    acc[0][1] = wmma_bf16(a0, b1, acc[0][1]);
    acc[1][1] = wmma_bf16(a1, b1, acc[1][1]);
    acc[0][2] = wmma_bf16(a0, b2, acc[0][2]);
    acc[1][2] = wmma_bf16(a1, b2, acc[1][2]);
    acc[0][3] = wmma_bf16(a0, b3, acc[0][3]);
    acc[1][3] = wmma_bf16(a1, b3, acc[1][3]);
  }
  const int col = nt * 16 + (lane & 15);
  const float b0 = bih[col] + bhh[col];
  const float b1 = bih[HE_ + col] + bhh[HE_ + col];
  const float b2 = bih[2 * HE_ + col] + bhh[2 * HE_ + col];
  const float b3 = bih[3 * HE_ + col] + bhh[3 * HE_ + col];
  const float w0 = Wih0[col], w1 = Wih0[HE_ + col], w2 = Wih0[2 * HE_ + col], w3 = Wih0[3 * HE_ + col];
#pragma unroll
  for (int m = 0; m < 2; ++m) {
#pragma unroll
    for (int i = 0; i < 8; ++i) {
      const size_t row = (size_t)mb * 32 + m * 16 + i + ((lane >> 4) << 3);
      float aval;
      if (s == 0) {
        aval = bos[0];
      } else {
        const int b = (int)(row / S_);
        const int ii = (int)(row % S_);
        const int hv = (ii == 0) ? 0 : heads[b * T_ + ii - 1];
        aval = (hv == s - 1) ? 1.0f : 0.0f;
      }
      const float gi = sigf(acc[m][0][i] + aval * w0 + b0);
      const float gf = sigf(acc[m][1][i] + aval * w1 + b1);
      const float gg = tanhf(acc[m][2][i] + aval * w2 + b2);
      const float go = sigf(acc[m][3][i] + aval * w3 + b3);
      const size_t ci = row * HE_ + col;
      const float c = gf * cbuf[ci] + gi * gg;
      cbuf[ci] = c;
      hnext[ci] = (bf16_t)(go * tanhf(c));
    }
  }
}

// ---------------- edge LSTM layer 1 step + fused cls reduction ----------------
// grid = (BS/32), block 128 (4 waves; wave w owns hidden cols [16w,16w+16)).
__global__ void edge1_step_kernel(const bf16_t* __restrict__ x, const bf16_t* __restrict__ h1prev,
                                  bf16_t* __restrict__ h1next, float* __restrict__ cbuf,
                                  const bf16_t* __restrict__ Wih1Sw, const bf16_t* __restrict__ Whh1Sw,
                                  const float* __restrict__ bih, const float* __restrict__ bhh,
                                  const float* __restrict__ clsW, const float* __restrict__ clsb,
                                  float* __restrict__ arc, int s) {
  __shared__ float hs[32][68];
  const int lane = threadIdx.x & 31;
  const int w = threadIdx.x >> 5;   // 0..3
  const int mb = blockIdx.x;        // 0..1023
  v8f acc[2][4];
#pragma unroll
  for (int m = 0; m < 2; ++m)
#pragma unroll
    for (int g = 0; g < 4; ++g) acc[m][g] = (v8f){};

  const bf16_t* X0 = x + (size_t)mb * 32 * HE_;
  const bf16_t* X1 = X0 + (size_t)16 * HE_;
  const bf16_t* H0r = h1prev + (size_t)mb * 32 * HE_;
  const bf16_t* H1r = H0r + (size_t)16 * HE_;

#pragma unroll
  for (int kt = 0; kt < 2; ++kt) {
    v16bf a0 = load_a_frag(X0, HE_, kt * 32, lane);
    v16bf a1 = load_a_frag(X1, HE_, kt * 32, lane);
    v16bf b0 = load_w_frag(Wih1Sw, (w + 0) * 2 + kt, lane);
    v16bf b1 = load_w_frag(Wih1Sw, (w + 4) * 2 + kt, lane);
    v16bf b2 = load_w_frag(Wih1Sw, (w + 8) * 2 + kt, lane);
    v16bf b3 = load_w_frag(Wih1Sw, (w + 12) * 2 + kt, lane);
    acc[0][0] = wmma_bf16(a0, b0, acc[0][0]);
    acc[1][0] = wmma_bf16(a1, b0, acc[1][0]);
    acc[0][1] = wmma_bf16(a0, b1, acc[0][1]);
    acc[1][1] = wmma_bf16(a1, b1, acc[1][1]);
    acc[0][2] = wmma_bf16(a0, b2, acc[0][2]);
    acc[1][2] = wmma_bf16(a1, b2, acc[1][2]);
    acc[0][3] = wmma_bf16(a0, b3, acc[0][3]);
    acc[1][3] = wmma_bf16(a1, b3, acc[1][3]);
  }
#pragma unroll
  for (int kt = 0; kt < 2; ++kt) {
    v16bf a0 = load_a_frag(H0r, HE_, kt * 32, lane);
    v16bf a1 = load_a_frag(H1r, HE_, kt * 32, lane);
    v16bf b0 = load_w_frag(Whh1Sw, (w + 0) * 2 + kt, lane);
    v16bf b1 = load_w_frag(Whh1Sw, (w + 4) * 2 + kt, lane);
    v16bf b2 = load_w_frag(Whh1Sw, (w + 8) * 2 + kt, lane);
    v16bf b3 = load_w_frag(Whh1Sw, (w + 12) * 2 + kt, lane);
    acc[0][0] = wmma_bf16(a0, b0, acc[0][0]);
    acc[1][0] = wmma_bf16(a1, b0, acc[1][0]);
    acc[0][1] = wmma_bf16(a0, b1, acc[0][1]);
    acc[1][1] = wmma_bf16(a1, b1, acc[1][1]);
    acc[0][2] = wmma_bf16(a0, b2, acc[0][2]);
    acc[1][2] = wmma_bf16(a1, b2, acc[1][2]);
    acc[0][3] = wmma_bf16(a0, b3, acc[0][3]);
    acc[1][3] = wmma_bf16(a1, b3, acc[1][3]);
  }

  const int col = w * 16 + (lane & 15);
  const float b0 = bih[col] + bhh[col];
  const float b1 = bih[HE_ + col] + bhh[HE_ + col];
  const float b2 = bih[2 * HE_ + col] + bhh[2 * HE_ + col];
  const float b3 = bih[3 * HE_ + col] + bhh[3 * HE_ + col];
#pragma unroll
  for (int m = 0; m < 2; ++m) {
#pragma unroll
    for (int i = 0; i < 8; ++i) {
      const int r = m * 16 + i + ((lane >> 4) << 3);
      const size_t row = (size_t)mb * 32 + r;
      const float gi = sigf(acc[m][0][i] + b0);
      const float gf = sigf(acc[m][1][i] + b1);
      const float gg = tanhf(acc[m][2][i] + b2);
      const float go = sigf(acc[m][3][i] + b3);
      const size_t ci = row * HE_ + col;
      const float c = gf * cbuf[ci] + gi * gg;
      cbuf[ci] = c;
      const float h = go * tanhf(c);
      h1next[ci] = (bf16_t)h;
      hs[r][col] = h;
    }
  }
  __syncthreads();
  if (s >= 1 && threadIdx.x < 32) {
    const int r = threadIdx.x;
    const size_t row = (size_t)mb * 32 + r;
    float a = clsb[0];
#pragma unroll
    for (int j = 0; j < HE_; ++j) a += hs[r][j] * clsW[j];
    arc[row * S_ + (s - 1)] = a;   // arc_logits[b, i, s-1], row = b*S+i
  }
}

// ---------------- host ----------------

extern "C" void kernel_launch(void* const* d_in, const int* in_sizes, int n_in,
                              void* d_out, int out_size, void* d_ws, size_t ws_size,
                              hipStream_t stream) {
  const float* inp   = (const float*)d_in[0];
  const float* tag   = (const float*)d_in[1];
  const int*   maskp = (const int*)d_in[2];
  const int*   heads = (const int*)d_in[3];
  const float* sent  = (const float*)d_in[4];
  const float* bos   = (const float*)d_in[5];
  const float* gWih0 = (const float*)d_in[6],  *gWhh0 = (const float*)d_in[7];
  const float* gbih0 = (const float*)d_in[8],  *gbhh0 = (const float*)d_in[9];
  const float* gWih1 = (const float*)d_in[10], *gWhh1 = (const float*)d_in[11];
  const float* gbih1 = (const float*)d_in[12], *gbhh1 = (const float*)d_in[13];
  const float* eWih0 = (const float*)d_in[14], *eWhh0 = (const float*)d_in[15];
  const float* ebih0 = (const float*)d_in[16], *ebhh0 = (const float*)d_in[17];
  const float* eWih1 = (const float*)d_in[18], *eWhh1 = (const float*)d_in[19];
  const float* ebih1 = (const float*)d_in[20], *ebhh1 = (const float*)d_in[21];
  const float* g2eW  = (const float*)d_in[22], *g2eb  = (const float*)d_in[23];
  const float* clsW  = (const float*)d_in[24], *clsb  = (const float*)d_in[25];
  const float* htW   = (const float*)d_in[26], *htb   = (const float*)d_in[27];
  const float* dtW   = (const float*)d_in[28], *dtb   = (const float*)d_in[29];

  float* out   = (float*)d_out;
  float* arc   = out;                                   // [B,S,S]
  float* headt = out + (size_t)B_ * S_ * S_;            // [B,S,100]
  float* dept  = headt + (size_t)B_ * S_ * DT_;         // [B,S,100]

  char* ws = (char*)d_ws;
  size_t off = 0;
  auto alloc = [&](size_t bytes) -> char* {
    char* p = ws + off;
    off = (off + bytes + 255) & ~(size_t)255;
    return p;
  };

  bf16_t* xbf  = (bf16_t*)alloc((size_t)B_ * S_ * DP_ * 2);
  bf16_t* H0   = (bf16_t*)alloc((size_t)B_ * S_ * HG_ * 2);
  bf16_t* Gbf  = (bf16_t*)alloc((size_t)B_ * S_ * HG_ * 2);
  auto wsz = [](int Npad, int KT) { return (size_t)(Npad / 16) * KT * 512 * 2; };
  bf16_t* wgih0 = (bf16_t*)alloc(wsz(1536, 16));
  bf16_t* wghh0 = (bf16_t*)alloc(wsz(1536, 12));
  bf16_t* wgih1 = (bf16_t*)alloc(wsz(1536, 12));
  bf16_t* wghh1 = (bf16_t*)alloc(wsz(1536, 12));
  bf16_t* wehh0 = (bf16_t*)alloc(wsz(256, 2));
  bf16_t* weih1 = (bf16_t*)alloc(wsz(256, 2));
  bf16_t* wehh1 = (bf16_t*)alloc(wsz(256, 2));
  bf16_t* wg2e  = (bf16_t*)alloc(wsz(64, 12));
  bf16_t* wht   = (bf16_t*)alloc(wsz(DTP_, 12));
  bf16_t* wdt   = (bf16_t*)alloc(wsz(DTP_, 12));
  bf16_t* hgA = (bf16_t*)alloc((size_t)B_ * HG_ * 2);
  bf16_t* hgB = (bf16_t*)alloc((size_t)B_ * HG_ * 2);
  float*  cg  = (float*)alloc((size_t)B_ * HG_ * 4);
  bf16_t* he0[2] = { (bf16_t*)alloc((size_t)BS_ * HE_ * 2), (bf16_t*)alloc((size_t)BS_ * HE_ * 2) };
  bf16_t* he1[2] = { (bf16_t*)alloc((size_t)BS_ * HE_ * 2), (bf16_t*)alloc((size_t)BS_ * HE_ * 2) };
  float*  ce0 = (float*)alloc((size_t)BS_ * HE_ * 4);
  float*  ce1 = (float*)alloc((size_t)BS_ * HE_ * 4);
  (void)ws_size; (void)in_sizes; (void)n_in;

  hipMemsetAsync(d_out, 0, (size_t)out_size * 4, stream);

  {
    size_t total = (size_t)B_ * S_ * DP_;
    build_x_kernel<<<dim3((unsigned)((total + 255) / 256)), 256, 0, stream>>>(inp, tag, sent, xbf);
  }
  auto swz = [&](const float* W, bf16_t* o, int N, int Npad, int K, int KT) {
    size_t total = (size_t)(Npad / 16) * KT * 512;
    swizzle_w_kernel<<<dim3((unsigned)((total + 255) / 256)), 256, 0, stream>>>(W, o, N, Npad, K, KT);
  };
  swz(gWih0, wgih0, 1536, 1536, D_, 16);
  swz(gWhh0, wghh0, 1536, 1536, HG_, 12);
  swz(gWih1, wgih1, 1536, 1536, HG_, 12);
  swz(gWhh1, wghh1, 1536, 1536, HG_, 12);
  swz(eWhh0, wehh0, 256, 256, HE_, 2);
  swz(eWih1, weih1, 256, 256, HE_, 2);
  swz(eWhh1, wehh1, 256, 256, HE_, 2);
  swz(g2eW,  wg2e,  HE_, HE_, HG_, 12);
  swz(htW,   wht,   DT_, DTP_, HG_, 12);
  swz(dtW,   wdt,   DT_, DTP_, HG_, 12);

  const dim3 gGrid(NTG_, 2);   // (24, 2)

  // ---- graph LSTM layer 0 ----
  hipMemsetAsync(hgA, 0, (size_t)B_ * HG_ * 2, stream);
  hipMemsetAsync(cg, 0, (size_t)B_ * HG_ * 4, stream);
  for (int t = 0; t < S_; ++t) {
    bf16_t* hp = (t & 1) ? hgB : hgA;
    bf16_t* hn = (t & 1) ? hgA : hgB;
    graph_step_kernel<16><<<gGrid, 32, 0, stream>>>(
        xbf + (size_t)t * DP_, S_ * DP_,
        wgih0, wghh0, gbih0, gbhh0,
        hp, hn, cg,
        H0 + (size_t)t * HG_, S_ * HG_, nullptr, t);
  }
  // ---- graph LSTM layer 1 (masked graph_state -> Gbf) ----
  hipMemsetAsync(hgA, 0, (size_t)B_ * HG_ * 2, stream);
  hipMemsetAsync(cg, 0, (size_t)B_ * HG_ * 4, stream);
  for (int t = 0; t < S_; ++t) {
    bf16_t* hp = (t & 1) ? hgB : hgA;
    bf16_t* hn = (t & 1) ? hgA : hgB;
    graph_step_kernel<12><<<gGrid, 32, 0, stream>>>(
        H0 + (size_t)t * HG_, S_ * HG_,
        wgih1, wghh1, gbih1, gbhh1,
        hp, hn, cg,
        Gbf + (size_t)t * HG_, S_ * HG_, maskp, t);
  }

  // ---- g2e projection: edge h-state init, zero c ----
  proj_kernel<<<dim3(HE_ / 16, BS_ / 32), 32, 0, stream>>>(Gbf, wg2e, g2eb, he0[0], he1[0], ce0, ce1);

  // ---- tag heads ----
  tag_kernel<<<dim3(DTP_ / 16, BS_ / 32), 32, 0, stream>>>(Gbf, wht, htb, headt);
  tag_kernel<<<dim3(DTP_ / 16, BS_ / 32), 32, 0, stream>>>(Gbf, wdt, dtb, dept);

  // ---- edge LSTM: 33 steps (step 0 = bos; steps 1..32 write band cols 0..31) ----
  for (int s = 0; s <= 32; ++s) {
    bf16_t* h0p = he0[s & 1];
    bf16_t* h0n = he0[(s + 1) & 1];
    bf16_t* h1p = he1[s & 1];
    bf16_t* h1n = he1[(s + 1) & 1];
    edge0_step_kernel<<<dim3(HE_ / 16, BS_ / 32), 32, 0, stream>>>(
        h0p, h0n, ce0, eWih0, ebih0, ebhh0, wehh0, heads, bos, s);
    edge1_step_kernel<<<dim3(BS_ / 32), 128, 0, stream>>>(
        h0n, h1p, h1n, ce1, weih1, wehh1, ebih1, ebhh1, clsW, clsb, arc, s);
  }
}
